// SPLoPAdapter_49813030699581
// MI455X (gfx1250) — matrix operand order repaired and checked by reading, not compiled, not run on hardware
//
#include <hip/hip_runtime.h>

typedef __attribute__((ext_vector_type(16))) _Float16 v16h;
typedef __attribute__((ext_vector_type(8)))  _Float16 v8h;
typedef __attribute__((ext_vector_type(8)))  float    v8f;

#define K_DIM 64
#define IJ_DIM 64          // I = n/p = 64, J = m/q = 64
#define P_DIM 32
#define Q_DIM 32
#define N_COLS 2048
#define S_SCALE 4096.0f    // 2^12, exact; lifts pos out of f16 denormal range
#define S_INV_SCALE (1.0f / 4096.0f)

// ---------------------------------------------------------------------------
// Prep: build f16 operands in workspace.
//   Sh[m][k]  = pos[k, m/64, m%64] * 2^12     (m = i*64+j,  4096 x 64)
//   Pt[n][k]  = sum_r pw[k,a,r]*ph[k,r,b]     (n = a*32+b,  1024 x 64)
// Both stored K-contiguous so the WMMA lane loads below are contiguous.
// ---------------------------------------------------------------------------
__global__ void splopa_prep(const float* __restrict__ pos,
                            const float* __restrict__ pw,
                            const float* __restrict__ ph,
                            int rdim,
                            _Float16* __restrict__ Sh,
                            _Float16* __restrict__ Pt)
{
    const int S_ELEMS = IJ_DIM * IJ_DIM * K_DIM;   // 262144
    const int P_ELEMS = P_DIM * Q_DIM * K_DIM;     // 65536
    int tid = blockIdx.x * blockDim.x + threadIdx.x;
    if (tid < S_ELEMS) {
        int m = tid >> 6;
        int k = tid & 63;
        float v = pos[k * (IJ_DIM * IJ_DIM) + m];
        Sh[m * K_DIM + k] = (_Float16)(v * S_SCALE);
    } else if (tid < S_ELEMS + P_ELEMS) {
        int t = tid - S_ELEMS;
        int n = t >> 6;            // a*32 + b
        int k = t & 63;
        int a = n >> 5;
        int b = n & 31;
        float acc = 0.0f;
        for (int rr = 0; rr < rdim; ++rr)
            acc += pw[(k * P_DIM + a) * rdim + rr] * ph[(k * rdim + rr) * Q_DIM + b];
        Pt[n * K_DIM + k] = (_Float16)acc;
    }
}

// ---------------------------------------------------------------------------
// Main: Delta = S @ Pt^T via v_wmma_f32_16x16x32_f16 (2 K-steps of 32),
// then out = weights + Delta * 2^-12, streamed once (HBM-bound, ~33.5 MB).
// One wave per 16x16 tile: 256 M-tiles x 64 N-tiles = 16384 waves.
// Tile -> output mapping: row_out = i*32 + a is CONSTANT per wave; each
// (r,hi) pair writes 16 contiguous floats (64B) in that row.
// ---------------------------------------------------------------------------
__global__ __launch_bounds__(256)
void splopa_main(const float* __restrict__ weights,
                 const _Float16* __restrict__ Sh,
                 const _Float16* __restrict__ Pt,
                 float* __restrict__ out)
{
    const int lane  = threadIdx.x;                 // 0..31
    const int wave  = threadIdx.y;                 // 0..7
    const int ntile = blockIdx.x;                  // 0..63
    const int mtile = blockIdx.y * 8 + wave;       // 0..255

    const int m0 = mtile * 16;
    const int n0 = ntile * 16;
    const int l2 = lane & 15;
    const int hi = lane >> 4;

    // A lane row: M = m0 + l2 ; B lane col: N = n0 + l2 (K-contiguous rows)
    const _Float16* __restrict__ arow = Sh + (m0 + l2) * K_DIM;
    const _Float16* __restrict__ brow = Pt + (n0 + l2) * K_DIM;

    v8f c = {};
#pragma unroll
    for (int kk = 0; kk < 2; ++kk) {
        const int kbase = kk * 32;
        // A 16x32 f16: lane<16 -> K 0..15; lane>=16 -> K 16..31 (M = l2)
        v16h a = *(const v16h*)(arow + kbase + hi * 16);
        // B 32x16 f16: lane<16 -> K 0..15; lane>=16 -> K 16..31 (N = l2)
        v16h b = *(const v16h*)(brow + kbase + hi * 16);

        c = __builtin_amdgcn_wmma_f32_16x16x32_f16(
                /*neg_a=*/false, a, /*neg_b=*/false, b,
                /*c_mod=*/(short)0, c, /*reuse_a=*/false, /*reuse_b=*/false);
    }

    // Delta[m,n] -> out[(i*32+a)*2048 + j*32+b], m=i*64+j, n=a*32+b
    const int i_blk  = m0 >> 6;
    const int j0     = m0 & 63;
    const int a_idx  = n0 >> 5;
    const int b0     = n0 & 31;
    const int rowoff = (i_blk * P_DIM + a_idx) * N_COLS;

#pragma unroll
    for (int r = 0; r < 8; ++r) {
        // C/D layout: VGPR r holds M = r + 8*hi (lanes split), N = l2
        const int col = (j0 + r + 8 * hi) * Q_DIM + b0 + l2;
        const int idx = rowoff + col;
        out[idx] = weights[idx] + c[r] * S_INV_SCALE;
    }
}

// ---------------------------------------------------------------------------
extern "C" void kernel_launch(void* const* d_in, const int* in_sizes, int n_in,
                              void* d_out, int out_size, void* d_ws, size_t ws_size,
                              hipStream_t stream)
{
    const float* weights = (const float*)d_in[0];
    const float* pos     = (const float*)d_in[1];
    const float* pw      = (const float*)d_in[2];
    const float* ph      = (const float*)d_in[3];
    float* out           = (float*)d_out;

    const int rdim = in_sizes[2] / (K_DIM * P_DIM);   // low-rank r (1 in setup)

    _Float16* Sh = (_Float16*)d_ws;                                   // 512 KB
    _Float16* Pt = (_Float16*)((char*)d_ws +
                   (size_t)IJ_DIM * IJ_DIM * K_DIM * sizeof(_Float16)); // +128 KB

    const int total = IJ_DIM * IJ_DIM * K_DIM + P_DIM * Q_DIM * K_DIM; // 327680
    splopa_prep<<<(total + 255) / 256, 256, 0, stream>>>(pos, pw, ph, rdim, Sh, Pt);

    dim3 block(32, 8);   // 8 waves of 32 (wave32)
    dim3 grid(64, 32);   // x: N-tiles, y: groups of 8 M-tiles
    splopa_main<<<grid, block, 0, stream>>>(weights, Sh, Pt, out);
}
